// C_EDESC_61632780698346
// MI455X (gfx1250) — matrix operand order repaired and checked by reading, not compiled, not run on hardware
//
#include <hip/hip_runtime.h>
#include <hip/hip_bf16.h>

// ---------------------------------------------------------------------------
// EDESC forward on MI455X (gfx1250, wave32, WMMA):
//   z = x@enc_W + enc_b ; x_bar = z@dec_W + dec_b ;
//   s = rownorm((sum_d (z@D)^2 + eta*d)/((eta+1)*d)) ; logits = z@pc_W + pc_b
//
// Split-precision f16 WMMA (hi/lo, 3 WMMAs ~ f32 accuracy), f32 accumulate.
//   GEMM1: z = x @ enc_W            [32768,204]x[204,1568]
//   GEMM2: P = z @ [dec_W|D|pc_W]^T [32768,1568]x[1568,300]  (z read once)
// Loaders split f32->f16(hi/lo) ONCE per element and store LDS tiles in
// per-lane WMMA fragment order => consumer path is ds_load_b128 + v_wmma only.
// ---------------------------------------------------------------------------

typedef __attribute__((ext_vector_type(16))) _Float16 v16h;
typedef __attribute__((ext_vector_type(2)))  _Float16 v2h;
typedef __attribute__((ext_vector_type(8)))  float    v8f;

#define N_ROWS   32768
#define N_INPUT  204
#define LATENT   1568
#define KCLUST   16
#define DSUB     5
#define NCAT     300          // 204 + 80 + 16
#define NCAT_PAD 304
#define ETA      5.0f

#define STG 68                // f32 staging row stride (conflict-free)

// Build Wcat^T [NCAT][LATENT]: row n is column n of the concatenated B.
__global__ __launch_bounds__(256) void prep_wcat_kernel(
    const float* __restrict__ dec_W,   // [LATENT, 204]
    const float* __restrict__ Dm,      // [LATENT, 80]
    const float* __restrict__ pc_W,    // [LATENT, 16]
    float* __restrict__ Wt)            // [NCAT, LATENT]
{
    int idx = blockIdx.x * 256 + threadIdx.x;
    if (idx >= NCAT * LATENT) return;
    int n = idx / LATENT;
    int k = idx - n * LATENT;
    float v;
    if (n < 204)       v = dec_W[(long)k * 204 + n];
    else if (n < 284)  v = Dm[(long)k * 80 + (n - 204)];
    else               v = pc_W[(long)k * 16 + (n - 284)];
    Wt[(long)n * LATENT + k] = v;
}

__device__ __forceinline__ void split2(float f0, float f1, v2h& hh, v2h& ll) {
    _Float16 h0 = (_Float16)f0, h1 = (_Float16)f1;
    hh[0] = h0; hh[1] = h1;
    ll[0] = (_Float16)(f0 - (float)h0);
    ll[1] = (_Float16)(f1 - (float)h1);
}

// Split-precision f16 WMMA GEMM: C[M,Ncols] = A[M,Kact] * B (+bias)
//   BT=false: B row-major [Kact, ldb]      (B[k][n] = B[k*ldb + n])
//   BT=true : B^T row-major [Ncols, ldb]   (B[k][n] = B[n*ldb + k])
// Block tile: 128(M) x 64(N), K-step 32, 8 wave32s; wave w -> rows w*16..+15.
// LDS f16 tiles are stored in per-lane fragment order:
//   A: frag(wv,lane)[j], lane = m + 16*k[3], j = 2*v+k[0],
//      v = 4*k[4] + (k>>1 & 3)                      (ISA 16-bit A 16x32 layout)
//   B: frag(nt,lane)[j], lane = n + 16*k[3], j = 8*k[4] + (k & 7)  (B 32x16 layout)
template <bool BT>
__global__ __launch_bounds__(256) void gemm_split_f16_kernel(
    const float* __restrict__ A, int lda, int Kact, int Ktiles,
    const float* __restrict__ B, int ldb, int Ncols,
    const float* __restrict__ bias,
    float* __restrict__ C, int ldc)
{
    __shared__ __align__(32) _Float16 sAh[8 * 32 * 16];   // 8 KB
    __shared__ __align__(32) _Float16 sAl[8 * 32 * 16];   // 8 KB
    __shared__ __align__(32) _Float16 sBh[4 * 32 * 16];   // 4 KB
    __shared__ __align__(32) _Float16 sBl[4 * 32 * 16];   // 4 KB
    __shared__ float stage[128 * STG];                    // ~34 KB

    const int t    = threadIdx.x;
    const int wave = t >> 5;
    const int lane = t & 31;
    const long m0  = (long)blockIdx.y * 128;
    const int  n0  = blockIdx.x * 64;

    v8f acc[4];
#pragma unroll
    for (int nt = 0; nt < 4; ++nt)
#pragma unroll
        for (int r = 0; r < 8; ++r) acc[nt][r] = 0.0f;

    for (int kt = 0; kt < Ktiles; ++kt) {
        const int k0 = kt * 32;

        // ---- stage A tile [128 x 32] as split f16 in fragment order ----
        // 2048 k-pairs, 8 per thread; float2 global reads, v2h LDS writes.
#pragma unroll
        for (int i = 0; i < 8; ++i) {
            int p = i * 256 + t;
            int r = p >> 4;
            int c = (p & 15) * 2;
            int gk = k0 + c;
            float f0 = 0.0f, f1 = 0.0f;
            if (gk < Kact) {            // Kact even -> pair-safe guard
                float2 f2 = *(const float2*)&A[(m0 + r) * lda + gk];
                f0 = f2.x; f1 = f2.y;
            }
            v2h hh, ll;
            split2(f0, f1, hh, ll);
            int fl   = (r & 15) + ((c & 8) ? 16 : 0);
            int wv   = r >> 4;
            int v    = ((c & 16) >> 2) | ((c >> 1) & 3);
            int base = (wv * 32 + fl) * 16 + 2 * v;
            *(v2h*)&sAh[base] = hh;
            *(v2h*)&sAl[base] = ll;
        }
        // ---- stage B tile [32 x 64] as split f16 in fragment order ----
        // 1024 k-pairs, 4 per thread.
#pragma unroll
        for (int i = 0; i < 4; ++i) {
            int p = i * 256 + t;
            int n, k;
            float f0 = 0.0f, f1 = 0.0f;
            if (!BT) {
                n = p & 63; k = (p >> 6) * 2;           // coalesced over n
                int gk = k0 + k, gn = n0 + n;
                if (gn < Ncols && gk < Kact) {
                    f0 = B[(long)gk * ldb + gn];
                    f1 = B[(long)(gk + 1) * ldb + gn];
                }
            } else {
                n = p >> 4; k = (p & 15) * 2;           // contiguous float2 in k
                int gk = k0 + k, gn = n0 + n;
                if (gn < Ncols && gk < Kact) {
                    float2 f2 = *(const float2*)&B[(long)gn * ldb + gk];
                    f0 = f2.x; f1 = f2.y;
                }
            }
            v2h hh, ll;
            split2(f0, f1, hh, ll);
            int fl   = (n & 15) + ((k & 8) ? 16 : 0);
            int nt   = n >> 4;
            int base = (nt * 32 + fl) * 16 + (((k & 16) >> 1) | (k & 7));
            *(v2h*)&sBh[base] = hh;
            *(v2h*)&sBl[base] = ll;
        }

        // prefetch next K-tile (gfx1250 global_prefetch_b8; speculative-safe)
        if (kt + 1 < Ktiles) {
            int nk = k0 + 32;
            if (t < 128) __builtin_prefetch(&A[(m0 + t) * lda + nk], 0, 0);
            if (!BT) { if (t < 32) __builtin_prefetch(&B[(long)(nk + t) * ldb + n0], 0, 0); }
            else     { if (t < 64) __builtin_prefetch(&B[(long)(n0 + t) * ldb + nk], 0, 0); }
        }

        __syncthreads();

        // ---- pure WMMA math path: v16h LDS loads (ds_load_b128 pairs) ----
        const int fb = (wave * 32 + lane) * 16;
        v16h ah = *(const v16h*)&sAh[fb];
        v16h al = *(const v16h*)&sAl[fb];
        v16h bh[4], bl[4];
#pragma unroll
        for (int nt = 0; nt < 4; ++nt) {
            int bb = (nt * 32 + lane) * 16;
            bh[nt] = *(const v16h*)&sBh[bb];
            bl[nt] = *(const v16h*)&sBl[bb];
        }
        // 3 rounds of 4 independent WMMAs (acc chains interleaved for ILP)
#pragma unroll
        for (int nt = 0; nt < 4; ++nt)
            acc[nt] = __builtin_amdgcn_wmma_f32_16x16x32_f16(
                false, ah, false, bh[nt], (short)0, acc[nt], false, false);
#pragma unroll
        for (int nt = 0; nt < 4; ++nt)
            acc[nt] = __builtin_amdgcn_wmma_f32_16x16x32_f16(
                false, ah, false, bl[nt], (short)0, acc[nt], false, false);
#pragma unroll
        for (int nt = 0; nt < 4; ++nt)
            acc[nt] = __builtin_amdgcn_wmma_f32_16x16x32_f16(
                false, al, false, bh[nt], (short)0, acc[nt], false, false);

        __syncthreads();
    }

    // ---- store via LDS staging: coalesced float4 global stores ----
    // C/D frag layout: col = lane&15, rows = r + 8*(lane>=16).
    {
        const int col = lane & 15;
        const int rb  = (lane >> 4) << 3;
#pragma unroll
        for (int nt = 0; nt < 4; ++nt)
#pragma unroll
            for (int r = 0; r < 8; ++r)
                stage[(wave * 16 + rb + r) * STG + nt * 16 + col] = acc[nt][r];
    }
    __syncthreads();
#pragma unroll
    for (int i = 0; i < 8; ++i) {
        int idx = i * 256 + t;
        int r  = idx >> 4;
        int c4 = (idx & 15) * 4;
        int gn = n0 + c4;
        if (gn < Ncols) {                      // Ncols % 4 == 0 -> vector-safe
            float4 v;
            v.x = stage[r * STG + c4 + 0];
            v.y = stage[r * STG + c4 + 1];
            v.z = stage[r * STG + c4 + 2];
            v.w = stage[r * STG + c4 + 3];
            if (bias) {
                v.x += bias[gn + 0]; v.y += bias[gn + 1];
                v.z += bias[gn + 2]; v.w += bias[gn + 3];
            }
            *(float4*)&C[(m0 + r) * ldc + gn] = v;
        }
    }
}

// Epilogue: one block per row of P[N, 304].
//   cols   0..203 -> x_bar (+dec_b)
//   cols 204..283 -> 16 groups of 5 -> squared sums -> soft assignment s
//   cols 284..299 -> logits (+pc_b)
__global__ __launch_bounds__(256) void epilogue_kernel(
    const float* __restrict__ P,
    const float* __restrict__ dec_b,
    const float* __restrict__ pc_b,
    float* __restrict__ xbar,
    float* __restrict__ s_out,
    float* __restrict__ logits)
{
    const long row = blockIdx.x;
    const float* p = P + row * NCAT_PAD;
    __shared__ float sk[KCLUST];
    const int t = threadIdx.x;

    for (int c = t; c < N_INPUT; c += 256)
        xbar[row * N_INPUT + c] = p[c] + dec_b[c];

    if (t < KCLUST) {
        float a = 0.0f;
#pragma unroll
        for (int dd = 0; dd < DSUB; ++dd) {
            float v = p[204 + t * DSUB + dd];
            a += v * v;
        }
        sk[t] = (a + ETA * (float)DSUB) / ((ETA + 1.0f) * (float)DSUB);
    }
    __syncthreads();
    if (t < KCLUST) {
        float tot = 0.0f;
#pragma unroll
        for (int i = 0; i < KCLUST; ++i) tot += sk[i];
        s_out[row * KCLUST + t]  = sk[t] / tot;
        logits[row * KCLUST + t] = p[284 + t] + pc_b[t];
    }
}

extern "C" void kernel_launch(void* const* d_in, const int* in_sizes, int n_in,
                              void* d_out, int out_size, void* d_ws, size_t ws_size,
                              hipStream_t stream) {
    const float* x     = (const float*)d_in[0];
    const float* enc_W = (const float*)d_in[1];
    const float* enc_b = (const float*)d_in[2];
    const float* dec_W = (const float*)d_in[3];
    const float* dec_b = (const float*)d_in[4];
    const float* Dm    = (const float*)d_in[5];
    const float* pc_W  = (const float*)d_in[6];
    const float* pc_b  = (const float*)d_in[7];

    float* out    = (float*)d_out;
    float* xbar   = out;                                        // [N, 204]
    float* s_out  = out + (size_t)N_ROWS * N_INPUT;             // [N, 16]
    float* z      = out + (size_t)N_ROWS * (N_INPUT + KCLUST);  // [N, 1568]
    float* logits = out + (size_t)N_ROWS * (N_INPUT + KCLUST + LATENT); // [N,16]

    float* Wcat = (float*)d_ws;                  // [300, 1568]  (~1.9 MB)
    float* P    = Wcat + (size_t)NCAT * LATENT;  // [N, 304]     (~40 MB)

    // 1) build concatenated B^T
    prep_wcat_kernel<<<(NCAT * LATENT + 255) / 256, 256, 0, stream>>>(
        dec_W, Dm, pc_W, Wcat);

    // 2) z = x @ enc_W + enc_b   (K=204 -> 7 K-tiles of 32, zero-padded)
    dim3 g1((LATENT + 63) / 64, N_ROWS / 128);
    gemm_split_f16_kernel<false><<<g1, 256, 0, stream>>>(
        x, N_INPUT, N_INPUT, 7,
        enc_W, LATENT, LATENT,
        enc_b, z, LATENT);

    // 3) P = z @ Wcat^T          (K=1568 -> 49 K-tiles)
    dim3 g2((NCAT + 63) / 64, N_ROWS / 128);
    gemm_split_f16_kernel<true><<<g2, 256, 0, stream>>>(
        z, LATENT, LATENT, 49,
        Wcat, LATENT, NCAT,
        nullptr, P, NCAT_PAD);

    // 4) epilogue: x_bar, s, logits
    epilogue_kernel<<<N_ROWS, 256, 0, stream>>>(
        P, dec_b, pc_b, xbar, s_out, logits);
}
